// NPGPLayer_84739704750365
// MI455X (gfx1250) — compile-verified, matrix operand order checked by reading, not run on hardware
//
#include <hip/hip_runtime.h>
#include <hip/hip_bf16.h>

// Problem constants (from reference)
#define S_  8
#define J_  4
#define D_  4
#define P_  3
#define N_  1024
#define BG_ 32
#define BU_ 32
#define MG_ 10
#define MU_ 32
#define PI_ 3.14159265358979323846f

typedef __attribute__((ext_vector_type(16))) _Float16 v16h;
typedef __attribute__((ext_vector_type(8)))  float    v8f;

// CDNA5 WMMA: D = A(16x32 f16) * B(32x16 f16) + C(16x16 f32)
__device__ __forceinline__ v8f wmma16x16x32(v16h a, v16h b, v8f c) {
  return __builtin_amdgcn_wmma_f32_16x16x32_f16(false, a, false, b, (short)0, c,
                                                false, false);
}

// 16-bit A/B matrix K index for vector element e, given lane group (hi = lane>=16).
// Per ISA 7.12.2: lanes 0-15 hold K=0..7,16..23 ; lanes 16-31 hold K=8..15,24..31.
__device__ __forceinline__ int kmap(int e, int hi) {
  return ((e < 8) ? e : (e + 8)) + (hi ? 8 : 0);
}

// ---------------------------------------------------------------------------
// Precompute 1: K[s,j,d,b] = cu[s,j,b] * prod_p Ghat[s,j,d,p,b]  (complex)
// (Ghat = RFF part + canonical part; fully scalar, 4096 threads.)
// ---------------------------------------------------------------------------
__global__ void k_Kcoef(const float* __restrict__ alphas, const float* __restrict__ pgs,
                        const float* __restrict__ zgs,    const float* __restrict__ thetag,
                        const float* __restrict__ betag,  const float* __restrict__ wg,
                        const float* __restrict__ qg,     const float* __restrict__ thetau,
                        const float* __restrict__ betau,  const float* __restrict__ wu,
                        float* __restrict__ Kre, float* __restrict__ Kim) {
  int idx = blockIdx.x * blockDim.x + threadIdx.x;
  if (idx >= S_ * J_ * D_ * BU_) return;
  int b = idx % BU_;
  int t = idx / BU_;
  int d = t % D_;
  int sj = t / D_;
  int j = sj % J_;
  int s = sj / J_;

  float Pre = 1.f, Pim = 0.f;
  for (int p = 0; p < P_; ++p) {
    float a   = alphas[d * P_ + p];
    float pg  = pgs[d * P_ + p];
    float om  = thetau[((s * J_ + j) * P_ + p) * BU_ + b];
    float i4a = 0.25f / a;
    float pref0 = 0.5f * sqrtf(PI_ / a);
    float Gre = 0.f, Gim = 0.f;
    for (int bg = 0; bg < BG_; ++bg) {
      float th = thetag[(d * P_ + p) * BG_ + bg];
      float e1 = __expf(-(th - om) * (th - om) * i4a);
      float e2 = __expf(-(th + om) * (th + om) * i4a);
      int   wb = ((s * D_ + d) * P_ + p) * BG_ + bg;
      float pr = pref0 * wg[wb];
      float be = betag[wb];
      Gre += pr * (e1 + e2) * __cosf(be);
      Gim += pr * (e1 - e2) * __sinf(be);
    }
    float A  = a + pg;
    float dec = __expf(-om * om * 0.25f / A);
    float cc  = sqrtf(PI_ / A);
    for (int g = 0; g < MG_; ++g) {
      float z  = zgs[g * P_ + p];
      float mu = pg * z / A;
      float rg = a * pg * z * z / A;
      float cf = cc * __expf(-rg) * dec;
      float q  = qg[((s * D_ + d) * P_ + p) * MG_ + g];
      float an = om * mu;             // exp(-i*om*mu)
      Gre += q * cf * __cosf(an);
      Gim -= q * cf * __sinf(an);
    }
    float nr = Pre * Gre - Pim * Gim;
    float ni = Pre * Gim + Pim * Gre;
    Pre = nr; Pim = ni;
  }
  int   sjb = (s * J_ + j) * BU_ + b;
  float wv  = wu[sjb], be = betau[sjb];
  float cre = wv * __cosf(be), cim = wv * __sinf(be);
  Kre[idx] = cre * Pre - cim * Pim;
  Kim[idx] = cre * Pim + cim * Pre;
}

// ---------------------------------------------------------------------------
// Precompute 2: coef1 = wg*cos(betag)*spec ; coef2n = -wg*sin(betag)*spec
// layout identical to wg: [s,d,p,bg]
// ---------------------------------------------------------------------------
__global__ void k_coef(const float* __restrict__ alphas, const float* __restrict__ pus,
                       const float* __restrict__ thetag, const float* __restrict__ betag,
                       const float* __restrict__ wg,
                       float* __restrict__ coef1, float* __restrict__ coef2n) {
  int idx = blockIdx.x * blockDim.x + threadIdx.x;
  if (idx >= S_ * D_ * P_ * BG_) return;
  int bg = idx % BG_;
  int t  = idx / BG_;
  int p  = t % P_;  t /= P_;
  int d  = t % D_;
  float a  = alphas[d * P_ + p];
  float Bm = a + pus[p];
  float th = thetag[(d * P_ + p) * BG_ + bg];
  float spec = sqrtf(PI_ / Bm) * __expf(-th * th * 0.25f / Bm);
  float w = wg[idx], be = betag[idx];
  coef1[idx]  =  w * __cosf(be) * spec;
  coef2n[idx] = -w * __sinf(be) * spec;
}

// ---------------------------------------------------------------------------
// Precompute 3: mp[d,p,n,m], exprp[d,p,n,m] = exp(-rp)
// ---------------------------------------------------------------------------
__global__ void k_mp(const float* __restrict__ ts, const float* __restrict__ zus,
                     const float* __restrict__ alphas, const float* __restrict__ pus,
                     float* __restrict__ mp, float* __restrict__ exprp) {
  int idx = blockIdx.x * blockDim.x + threadIdx.x;
  if (idx >= D_ * P_ * N_ * MU_) return;
  int m = idx % MU_;
  int t = idx / MU_;
  int n = t % N_;  t /= N_;
  int p = t % P_;
  int d = t / P_;
  float a  = alphas[d * P_ + p];
  float pu = pus[p];
  float Bm = a + pu;
  float c  = ts[n * P_ + p] - zus[m * P_ + p];
  mp[idx]    = pu * c / Bm;
  exprp[idx] = __expf(-a * pu * c * c / Bm);
}

// ---------------------------------------------------------------------------
// Main kernel: per (d, 16-n tile) block of 256 threads (8 waves).
// Stage 1: Gg[s,p,nm] via 3 WMMAs per (row-tile,p); product over p in f32.
// Stage 2: F_can[s,j,d,n] via one WMMA per s (wave == s), prodg staged in LDS.
// All predication is done with branchless index-clamp + value-select so the
// wave stays converged (EXEC all-1s) through the WMMA pipeline.
// ---------------------------------------------------------------------------
__global__ __launch_bounds__(256)
void k_gg_fcan(const float* __restrict__ mp,    const float* __restrict__ exprp,
               const float* __restrict__ coef1, const float* __restrict__ coef2n,
               const float* __restrict__ qg,    const float* __restrict__ qu,
               const float* __restrict__ alphas,const float* __restrict__ pgs,
               const float* __restrict__ pus,   const float* __restrict__ zgs,
               const float* __restrict__ thetag,
               float* __restrict__ Fcan) {
  const int NMT = 512;                            // 16 n * 32 m rows per block
  __shared__ _Float16 prodg[S_ * 512];            // 8 KB

  const int d    = blockIdx.x / (N_ / 16);
  const int nt   = blockIdx.x % (N_ / 16);
  const int n0   = nt * 16;
  const int wave = threadIdx.x >> 5;
  const int lane = threadIdx.x & 31;
  const int hi   = (lane & 16) ? 1 : 0;
  const int col  = lane & 15;                     // WMMA B/D column
  const int rowA = lane & 15;                     // WMMA A row (within tile)
  const int csS  = col & (S_ - 1);                // clamped column for s-indexed loads
  const bool okS = (col < S_);
  const int csJ  = col & (J_ - 1);                // clamped column for j-indexed loads
  const bool okJ = (col < J_);

  v8f prod[4];
#pragma unroll
  for (int i = 0; i < 4; ++i)
#pragma unroll
    for (int r = 0; r < 8; ++r) prod[i][r] = 1.0f;

  for (int p = 0; p < P_; ++p) {
    const int dp  = d * P_ + p;
    const float a  = alphas[dp];
    const float pg = pgs[dp];
    const float Bm = a + pus[p];
    const float Cm = Bm + pg;
    const float sqc = sqrtf(PI_ / Cm);
    const float bc  = Bm * pg / Cm;

    // B matrices for this (d,p): columns = s (8 used of 16), K = bg / g.
    // Branchless: always load from clamped (valid) index, then cndmask-select.
    v16h B1, B2, Bq;
    float thg[16]; float zg[16];
#pragma unroll
    for (int e = 0; e < 16; ++e) {
      int k  = kmap(e, hi);
      int kg = (k < MG_) ? k : 0;                 // clamped g index
      bool okG = (k < MG_);
      thg[e] = thetag[dp * BG_ + k];
      zg[e]  = zgs[kg * P_ + p];
      int base = (csS * D_ + d) * P_ + p;
      float b1 = coef1[base * BG_ + k];
      float b2 = coef2n[base * BG_ + k];
      float bq = qg[base * MG_ + kg];
      B1[e] = (_Float16)(okS ? b1 : 0.f);
      B2[e] = (_Float16)(okS ? b2 : 0.f);
      Bq[e] = (_Float16)((okS && okG) ? bq : 0.f);
    }

#pragma unroll
    for (int i = 0; i < 4; ++i) {
      const int rt  = wave * 4 + i;               // row tile 0..31
      const int row = rt * 16 + rowA;             // nm row for this lane's A row
      const int nl  = row >> 5, m = row & 31;
      const int idx = (dp * N_ + (n0 + nl)) * MU_ + m;
      const float mpr = mp[idx];
      const float er  = exprp[idx];

      v16h Ac, As, Av;
#pragma unroll
      for (int e = 0; e < 16; ++e) {
        int   k   = kmap(e, hi);
        float arg = thg[e] * mpr;
        Ac[e] = (_Float16)__cosf(arg);
        As[e] = (_Float16)__sinf(arg);
        float diff = mpr - zg[e];
        float vv = sqc * er * __expf(-bc * diff * diff);  // speculative, cheap (trans unit)
        Av[e] = (_Float16)((k < MG_) ? vv : 0.f);
      }

      v8f c = {};
      c = wmma16x16x32(Ac, B1, c);                // + cos . coef1
      c = wmma16x16x32(As, B2, c);                // - sin . coef2
      v8f cc = {};
      cc = wmma16x16x32(Av, Bq, cc);              // canonical term

#pragma unroll
      for (int r = 0; r < 8; ++r) {
        int rowD = rt * 16 + r + (hi ? 8 : 0);    // D-row layout differs from A
        int idxD = (dp * N_ + (n0 + (rowD >> 5))) * MU_ + (rowD & 31);
        float g  = c[r] * exprp[idxD] + cc[r];    // Gg = term_rff*exp(-rp)+term_can
        prod[i][r] *= g;
      }
    }
  }

  // Stage prodg to LDS: prodg[s][n_local*32 + m], f16.
  // Duplicate columns 8..15 harmlessly alias into 0..7 (same values would be
  // garbage) -> keep the store guard (stores may be EXEC-predicated freely).
#pragma unroll
  for (int i = 0; i < 4; ++i) {
    int rt = wave * 4 + i;
#pragma unroll
    for (int r = 0; r < 8; ++r) {
      int rowD = rt * 16 + r + (hi ? 8 : 0);
      if (okS) prodg[col * NMT + rowD] = (_Float16)prod[i][r];
    }
  }
  __syncthreads();

  // Stage 2: wave w handles s = w.  A[n_local, m] from LDS, B = qu[s, j, m].
  {
    const int s = wave;
    v16h A, Bq2;
#pragma unroll
    for (int e = 0; e < 16; ++e) {
      int k = kmap(e, hi);
      A[e]  = prodg[s * NMT + rowA * 32 + k];
      float q = qu[(s * J_ + csJ) * MU_ + k];
      Bq2[e] = (_Float16)(okJ ? q : 0.f);
    }
    v8f c = {};
    c = wmma16x16x32(A, Bq2, c);
#pragma unroll
    for (int r = 0; r < 8; ++r) {
      int n = n0 + r + (hi ? 8 : 0);
      if (okJ) Fcan[((s * J_ + col) * D_ + d) * N_ + n] = c[r];
    }
  }
}

// ---------------------------------------------------------------------------
// F_rff kernel: per (s, j, 16-n tile) wave tile.
// F_rff[s,j,d,n] = sum_b ReK*cos(phi) - ImK*sin(phi), phi = sum_p thetau*ts[n,p]
// ---------------------------------------------------------------------------
__global__ __launch_bounds__(256)
void k_frff(const float* __restrict__ ts, const float* __restrict__ thetau,
            const float* __restrict__ Kre, const float* __restrict__ Kim,
            float* __restrict__ Frff) {
  const int wave = threadIdx.x >> 5;
  const int lane = threadIdx.x & 31;
  const int hi   = (lane & 16) ? 1 : 0;
  const int col  = lane & 15;
  const int csD  = col & (D_ - 1);
  const bool okD = (col < D_);

  const int tile = blockIdx.x * 8 + wave;         // 2048 tiles
  const int s   = tile / (J_ * (N_ / 16));
  const int rem = tile % (J_ * (N_ / 16));
  const int j   = rem / (N_ / 16);
  const int nt  = rem % (N_ / 16);
  const int n0  = nt * 16;

  const int nA = n0 + (lane & 15);                // A row -> n
  const float t0 = ts[nA * P_ + 0];
  const float t1 = ts[nA * P_ + 1];
  const float t2 = ts[nA * P_ + 2];
  const float* tu = thetau + (size_t)((s * J_ + j) * P_) * BU_;

  v16h Ac, As, Bre, Bim;
#pragma unroll
  for (int e = 0; e < 16; ++e) {
    int   b   = kmap(e, hi);
    float phi = tu[0 * BU_ + b] * t0 + tu[1 * BU_ + b] * t1 + tu[2 * BU_ + b] * t2;
    Ac[e] = (_Float16)__cosf(phi);
    As[e] = (_Float16)__sinf(phi);
    int ki = ((s * J_ + j) * D_ + csD) * BU_ + b;
    float re = Kre[ki];
    float im = -Kim[ki];
    Bre[e] = (_Float16)(okD ? re : 0.f);
    Bim[e] = (_Float16)(okD ? im : 0.f);
  }
  v8f c = {};
  c = wmma16x16x32(Ac, Bre, c);
  c = wmma16x16x32(As, Bim, c);
#pragma unroll
  for (int r = 0; r < 8; ++r) {
    int n = n0 + r + (hi ? 8 : 0);
    if (okD) Frff[((s * J_ + j) * D_ + col) * N_ + n] = c[r];
  }
}

// ---------------------------------------------------------------------------
// Final: out[s,n,d] = sum_j amps[d,j]*ampu*(Frff+Fcan)[s,j,d,n] + (ts@W)[n,d]
// ---------------------------------------------------------------------------
__global__ void k_final(const float* __restrict__ Frff, const float* __restrict__ Fcan,
                        const float* __restrict__ log_amps, const float* __restrict__ ampu_p,
                        const float* __restrict__ ts, const float* __restrict__ W,
                        float* __restrict__ out) {
  int idx = blockIdx.x * blockDim.x + threadIdx.x;
  if (idx >= S_ * N_ * D_) return;
  int d = idx % D_;
  int t = idx / D_;
  int n = t % N_;
  int s = t / N_;
  float acc = 0.f;
#pragma unroll
  for (int j = 0; j < J_; ++j) {
    int fi = ((s * J_ + j) * D_ + d) * N_ + n;
    acc += __expf(log_amps[d * J_ + j]) * (Frff[fi] + Fcan[fi]);
  }
  acc *= ampu_p[0];
  float tw = 0.f;
#pragma unroll
  for (int p = 0; p < P_; ++p) tw += ts[n * P_ + p] * W[p * D_ + d];
  out[idx] = acc + tw;
}

// ---------------------------------------------------------------------------
extern "C" void kernel_launch(void* const* d_in, const int* in_sizes, int n_in,
                              void* d_out, int out_size, void* d_ws, size_t ws_size,
                              hipStream_t stream) {
  const float* ts       = (const float*)d_in[0];
  const float* alphas   = (const float*)d_in[1];
  const float* pgs      = (const float*)d_in[2];
  const float* zgs      = (const float*)d_in[3];
  const float* thetag   = (const float*)d_in[4];
  const float* betag    = (const float*)d_in[5];
  const float* wg       = (const float*)d_in[6];
  const float* qg       = (const float*)d_in[7];
  const float* pus      = (const float*)d_in[8];
  const float* zus      = (const float*)d_in[9];
  const float* thetau   = (const float*)d_in[10];
  const float* betau    = (const float*)d_in[11];
  const float* wu       = (const float*)d_in[12];
  const float* qu       = (const float*)d_in[13];
  const float* ampu     = (const float*)d_in[14];
  const float* log_amps = (const float*)d_in[15];
  const float* W        = (const float*)d_in[16];
  float* out = (float*)d_out;

  // Workspace layout (floats), ~4.3 MB total
  float* ws = (float*)d_ws;
  float* coef1  = ws;                     // 3072
  float* coef2n = coef1  + S_*D_*P_*BG_;  // 3072
  float* Kre    = coef2n + S_*D_*P_*BG_;  // 4096
  float* Kim    = Kre    + S_*J_*D_*BU_;  // 4096
  float* mp     = Kim    + S_*J_*D_*BU_;  // 393216
  float* exprp  = mp     + D_*P_*N_*MU_;  // 393216
  float* Frff   = exprp  + D_*P_*N_*MU_;  // 131072
  float* Fcan   = Frff   + S_*J_*D_*N_;   // 131072

  k_coef<<<(S_*D_*P_*BG_ + 255) / 256, 256, 0, stream>>>(
      alphas, pus, thetag, betag, wg, coef1, coef2n);

  k_Kcoef<<<(S_*J_*D_*BU_ + 255) / 256, 256, 0, stream>>>(
      alphas, pgs, zgs, thetag, betag, wg, qg, thetau, betau, wu, Kre, Kim);

  k_mp<<<(D_*P_*N_*MU_ + 255) / 256, 256, 0, stream>>>(
      ts, zus, alphas, pus, mp, exprp);

  k_gg_fcan<<<D_ * (N_ / 16), 256, 0, stream>>>(
      mp, exprp, coef1, coef2n, qg, qu, alphas, pgs, pus, zgs, thetag, Fcan);

  k_frff<<<(S_ * J_ * (N_ / 16)) / 8, 256, 0, stream>>>(
      ts, thetau, Kre, Kim, Frff);

  k_final<<<(S_*N_*D_ + 255) / 256, 256, 0, stream>>>(
      Frff, Fcan, log_amps, ampu, ts, W, out);
}
